// GNN_24567212933214
// MI455X (gfx1250) — compile-verified, matrix operand order checked by reading, not run on hardware
//
#include <hip/hip_runtime.h>
#include <hip/hip_bf16.h>
#include <math.h>

typedef __bf16 bf16_t;
typedef __attribute__((ext_vector_type(16))) __bf16 v16bf;
typedef __attribute__((ext_vector_type(8)))  float  v8f;

#define NNODE   32768
#define NEDGE   524288
#define FIN     128
#define DIM     64
#define NHEAD   3
#define HCDIM   192
#define NGRAPH  256
#define NPG     128
#define DNHID   256
#define NCLS    10

// ---------------- utility kernels ----------------

__global__ void k_fill_f32(float* p, float v, size_t n) {
    size_t i = (size_t)blockIdx.x * blockDim.x + threadIdx.x;
    if (i < n) p[i] = v;
}

__global__ void k_fill_u32(unsigned* p, unsigned v, size_t n) {
    size_t i = (size_t)blockIdx.x * blockDim.x + threadIdx.x;
    if (i < n) p[i] = v;
}

__global__ void k_cvt_f32_bf16(const float* __restrict__ in, bf16_t* __restrict__ out, size_t n) {
    size_t i = (size_t)blockIdx.x * blockDim.x + threadIdx.x;
    if (i < n) out[i] = (bf16_t)in[i];
}

// Convert weight [K, Nout] (fp32, row-major) -> transposed, padded bf16 [NoutPad, K].
// Rows (output columns) >= Nout are zero-filled so the GEMM needs no bounds checks.
__global__ void k_cvt_transpose_pad(const float* __restrict__ in, bf16_t* __restrict__ out,
                                    int K, int Nout, int NoutPad) {
    int idx = blockIdx.x * blockDim.x + threadIdx.x;
    if (idx >= NoutPad * K) return;
    int col = idx / K;     // output column = row of Wt
    int kk  = idx % K;
    out[idx] = (col < Nout) ? (bf16_t)in[(size_t)kk * Nout + col] : (bf16_t)0.0f;
}

// ---------------- WMMA GEMM: C[M,Nout] = epilogue(A[M,K] @ W[K,Nout] + bias) -----
// A bf16 row-major [M,K].  Wt bf16 TRANSPOSED+padded [NoutPad,K].
// One wave -> one 16x16 tile of C; 4 waves / block stacked along M (64 rows).
// Fragments are single aligned 256-bit loads (lanes 0-15: K lo half,
// lanes 16-31: K hi half).  K must be a multiple of 64 (two WMMAs / iter).
// mode: 0 = none, 1 = relu, 2 = relu + batchnorm(eval)

__global__ void k_wmma_gemm_bf16(const bf16_t* __restrict__ A,
                                 const bf16_t* __restrict__ Wt,
                                 const float* __restrict__ bias,
                                 float* __restrict__ C,
                                 int M, int K, int Nout, int ldc, int mode,
                                 const float* __restrict__ bng,
                                 const float* __restrict__ bnb,
                                 const float* __restrict__ bnm,
                                 const float* __restrict__ bnv) {
    const int wave    = threadIdx.x >> 5;
    const int lane    = threadIdx.x & 31;
    const int lrow    = lane & 15;
    const int halfOff = (lane >> 4) * 16;    // K offset for this lane half

    const int rowBase = blockIdx.x * 64 + wave * 16;
    const int colBase = blockIdx.y * 16;

    const bf16_t* __restrict__ arow = A  + (size_t)(rowBase + lrow) * K + halfOff;
    const bf16_t* __restrict__ brow = Wt + (size_t)(colBase + lrow) * K + halfOff;

    v8f acc = {};

    for (int k0 = 0; k0 < K; k0 += 64) {
        v16bf a0 = *(const v16bf*)(arow + k0);
        v16bf b0 = *(const v16bf*)(brow + k0);
        v16bf a1 = *(const v16bf*)(arow + k0 + 32);
        v16bf b1 = *(const v16bf*)(brow + k0 + 32);
        if (k0 + 64 < K) __builtin_prefetch(arow + k0 + 64, 0, 0);
        acc = __builtin_amdgcn_wmma_f32_16x16x32_bf16(false, a0, false, b0,
                                                      (short)0, acc, false, false);
        acc = __builtin_amdgcn_wmma_f32_16x16x32_bf16(false, a1, false, b1,
                                                      (short)0, acc, false, false);
    }

    // epilogue: C/D layout -> VGPR i: row = rowBase + i + 8*(lane>=16), col = lane%16
    const int col = colBase + lrow;
    if (col < Nout) {
        float bv = bias ? bias[col] : 0.0f;
        float gg = 1.f, bb = 0.f, mm = 0.f, vv = 1.f;
        if (mode == 2) { gg = bng[col]; bb = bnb[col]; mm = bnm[col]; vv = bnv[col]; }
        float rstd = (mode == 2) ? __frsqrt_rn(vv + 1e-5f) : 1.0f;
        const int rhalf = (lane >> 4) * 8;
#pragma unroll
        for (int i = 0; i < 8; ++i) {
            int row = rowBase + i + rhalf;
            float val = acc[i] + bv;
            if (mode >= 1) val = fmaxf(val, 0.0f);
            if (mode == 2) val = (val - mm) * rstd * gg + bb;
            if (row < M) C[(size_t)row * ldc + col] = val;
        }
    }
}

// ---------------- edge attention kernels ----------------

__device__ __forceinline__ unsigned f2ord(float f) {
    unsigned u = __float_as_uint(f);
    return (u & 0x80000000u) ? ~u : (u | 0x80000000u);
}
__device__ __forceinline__ float ord2f(unsigned u) {
    unsigned b = (u & 0x80000000u) ? (u ^ 0x80000000u) : ~u;
    return __uint_as_float(b);
}

__global__ void k_edge_logits(const float* __restrict__ q, const float* __restrict__ kk,
                              const int* __restrict__ src, const int* __restrict__ dst,
                              const float* __restrict__ mask,
                              float* __restrict__ alog, unsigned* __restrict__ amaxb) {
    int idx = blockIdx.x * blockDim.x + threadIdx.x;
    if (idx >= NEDGE * NHEAD) return;
    int e = idx / NHEAD, h = idx % NHEAD;
    int s = src[e], d = dst[e];
    bool valid = (mask[s] > 0.5f) && (mask[d] > 0.5f);
    float a = -1e9f;
    if (valid) {
        const float* qp = q  + (size_t)d * HCDIM + h * DIM;
        const float* kp = kk + (size_t)s * HCDIM + h * DIM;
        float acc = 0.f;
#pragma unroll 8
        for (int i = 0; i < DIM; ++i) acc = fmaf(qp[i], kp[i], acc);
        a = acc * 0.125f;  // 1/sqrt(64)
        atomicMax(&amaxb[d * NHEAD + h], f2ord(a));
    }
    alog[idx] = a;
}

__global__ void k_edge_softmax(const float* __restrict__ alog,
                               const unsigned* __restrict__ amaxb,
                               const int* __restrict__ dst,
                               float* __restrict__ p, float* __restrict__ den) {
    int idx = blockIdx.x * blockDim.x + threadIdx.x;
    if (idx >= NEDGE * NHEAD) return;
    int e = idx / NHEAD, h = idx % NHEAD;
    float a  = alog[idx];
    float pe = 0.f;
    if (a > -5e8f) {
        pe = __expf(a - ord2f(amaxb[dst[e] * NHEAD + h]));
        atomicAdd(&den[dst[e] * NHEAD + h], pe);
    }
    p[idx] = pe;
}

__global__ void k_edge_aggregate(const float* __restrict__ p, const float* __restrict__ den,
                                 const float* __restrict__ v,
                                 const int* __restrict__ src, const int* __restrict__ dst,
                                 float* __restrict__ outp) {
    int idx = blockIdx.x * blockDim.x + threadIdx.x;
    if (idx >= NEDGE * NHEAD) return;
    int e = idx / NHEAD, h = idx % NHEAD;
    float pe = p[idx];
    if (pe == 0.f) return;
    float w = pe / (den[dst[e] * NHEAD + h] + 1e-16f);
    const float* vp = v    + (size_t)src[e] * HCDIM + h * DIM;
    float*       op = outp + (size_t)dst[e] * HCDIM + h * DIM;
#pragma unroll 8
    for (int i = 0; i < DIM; ++i) atomicAdd(&op[i], w * vp[i]);
}

// beta-gated skip: beta = sigmoid([out,r,out-r]@Wbeta); x = beta*r + (1-beta)*out
__global__ void k_beta_gate(const float* __restrict__ agg, const float* __restrict__ r,
                            const float* __restrict__ Wbeta, bf16_t* __restrict__ xout) {
    int n = blockIdx.x * blockDim.x + threadIdx.x;
    if (n >= NNODE) return;
    const float* o  = agg + (size_t)n * HCDIM;
    const float* rr = r   + (size_t)n * HCDIM;
    float z = 0.f;
#pragma unroll 4
    for (int c = 0; c < HCDIM; ++c)
        z += o[c] * Wbeta[c] + rr[c] * Wbeta[HCDIM + c] + (o[c] - rr[c]) * Wbeta[2 * HCDIM + c];
    float beta = 1.f / (1.f + __expf(-z));
#pragma unroll 4
    for (int c = 0; c < HCDIM; ++c)
        xout[(size_t)n * HCDIM + c] = (bf16_t)(beta * rr[c] + (1.f - beta) * o[c]);
}

// ---------------- TopK pooling (one block per graph, 128 threads) -------------

__global__ void k_topk_pool(float* __restrict__ x, const float* __restrict__ w,
                            float* __restrict__ mask, int kkeep) {
    __shared__ float sc[NPG];
    __shared__ float wsh[DIM];
    __shared__ float nrm;
    int g = blockIdx.x, t = threadIdx.x;
    if (t < DIM) wsh[t] = w[t];
    __syncthreads();
    if (t == 0) {
        float s = 0.f;
        for (int i = 0; i < DIM; ++i) s += wsh[i] * wsh[i];
        nrm = rsqrtf(s);
    }
    __syncthreads();
    int n = g * NPG + t;
    float s = 0.f;
    for (int i = 0; i < DIM; ++i) s += x[(size_t)n * DIM + i] * wsh[i];
    s *= nrm;
    float sm = (mask[n] > 0.5f) ? s : -1e30f;
    sc[t] = sm;
    __syncthreads();
    int rank = 0;
    for (int j = 0; j < NPG; ++j) {
        float o = sc[j];
        rank += (o > sm) || (o == sm && j < t);
    }
    bool keep  = rank < kkeep;
    float gate = keep ? tanhf(s) : 0.f;
    for (int i = 0; i < DIM; ++i) x[(size_t)n * DIM + i] *= gate;
    mask[n] = keep ? 1.f : 0.f;
}

// mean||max global pool, accumulated into gacc[B, 2*DIM]
__global__ void k_global_pool(const float* __restrict__ x, const float* __restrict__ mask,
                              float* __restrict__ gacc, float invk) {
    int g = blockIdx.x, d = threadIdx.x;  // 64 threads
    float sum = 0.f, mx = -1e30f;
    for (int i = 0; i < NPG; ++i) {
        int n = g * NPG + i;
        if (mask[n] > 0.5f) {
            float v = x[(size_t)n * DIM + d];
            sum += v;
            mx = fmaxf(mx, v);
        }
    }
    gacc[(size_t)g * (2 * DIM) + d]       += sum * invk;
    gacc[(size_t)g * (2 * DIM) + DIM + d] += mx;
}

// ---------------- launch orchestration ----------------

extern "C" void kernel_launch(void* const* d_in, const int* in_sizes, int n_in,
                              void* d_out, int out_size, void* d_ws, size_t ws_size,
                              hipStream_t stream) {
    (void)in_sizes; (void)n_in; (void)out_size; (void)ws_size;

    // --- input pointers (pytree leaves in setup_inputs() insertion order) ---
    const float* x_in = (const float*)d_in[0];
    struct ConvW { const float *Wq,*bq,*Wk,*bk,*Wv,*bv,*Ws,*bs,*Wbeta; };
    auto conv_at = [&](int base) {
        ConvW c;
        c.Wq = (const float*)d_in[base+0]; c.bq = (const float*)d_in[base+1];
        c.Wk = (const float*)d_in[base+2]; c.bk = (const float*)d_in[base+3];
        c.Wv = (const float*)d_in[base+4]; c.bv = (const float*)d_in[base+5];
        c.Ws = (const float*)d_in[base+6]; c.bs = (const float*)d_in[base+7];
        c.Wbeta = (const float*)d_in[base+8];
        return c;
    };
    ConvW conv[5];
    conv[0] = conv_at(1);                       // conv1
    const float* transfW[5]; const float* transfB[5];
    transfW[0] = (const float*)d_in[10]; transfB[0] = (const float*)d_in[11];
    const float *bng[5], *bnb[5], *bnm[5], *bnv[5];
    bng[0] = (const float*)d_in[12]; bnb[0] = (const float*)d_in[13];
    bnm[0] = (const float*)d_in[14]; bnv[0] = (const float*)d_in[15];
    for (int i = 0; i < 4; ++i) conv[1 + i] = conv_at(16 + 9 * i);
    for (int i = 0; i < 4; ++i) { transfW[1+i] = (const float*)d_in[52 + 2*i];
                                  transfB[1+i] = (const float*)d_in[53 + 2*i]; }
    for (int i = 0; i < 4; ++i) { bng[1+i] = (const float*)d_in[60 + 4*i];
                                  bnb[1+i] = (const float*)d_in[61 + 4*i];
                                  bnm[1+i] = (const float*)d_in[62 + 4*i];
                                  bnv[1+i] = (const float*)d_in[63 + 4*i]; }
    const float* poolw[4];
    for (int i = 0; i < 4; ++i) poolw[i] = (const float*)d_in[76 + i];
    const float* headW[3] = {(const float*)d_in[80], (const float*)d_in[82], (const float*)d_in[84]};
    const float* headB[3] = {(const float*)d_in[81], (const float*)d_in[83], (const float*)d_in[85]};
    const int* edge_idx = (const int*)d_in[86];
    const int* e_src = edge_idx;
    const int* e_dst = edge_idx + NEDGE;

    // --- workspace bump allocator ---
    size_t off = 0;
    auto alloc = [&](size_t bytes) -> void* {
        void* p = (char*)d_ws + off;
        off += (bytes + 255) & ~(size_t)255;
        return p;
    };
    // bf16 transposed+padded weights
    bf16_t* cWb[5][4];
    int cdin[5] = {FIN, DIM, DIM, DIM, DIM};
    for (int l = 0; l < 5; ++l)
        for (int j = 0; j < 4; ++j)
            cWb[l][j] = (bf16_t*)alloc((size_t)HCDIM * cdin[l] * sizeof(bf16_t));
    bf16_t* tWb[5];
    for (int l = 0; l < 5; ++l) tWb[l] = (bf16_t*)alloc((size_t)DIM * HCDIM * sizeof(bf16_t));
    bf16_t* hWb[3];
    int hK[3] = {2 * DIM, DNHID, DNHID / 2};
    int hN[3] = {DNHID, DNHID / 2, NCLS};
    int hNp[3] = {DNHID, DNHID / 2, 16};        // padded to 16
    for (int j = 0; j < 3; ++j) hWb[j] = (bf16_t*)alloc((size_t)hNp[j] * hK[j] * sizeof(bf16_t));
    // activations / scratch
    bf16_t* xin_b16  = (bf16_t*)alloc((size_t)NNODE * FIN   * sizeof(bf16_t));
    bf16_t* xhc_b16  = (bf16_t*)alloc((size_t)NNODE * HCDIM * sizeof(bf16_t));
    bf16_t* x64_b16  = (bf16_t*)alloc((size_t)NNODE * DIM   * sizeof(bf16_t));
    float*  x64      = (float*) alloc((size_t)NNODE * DIM   * sizeof(float));
    float*  qbuf     = (float*) alloc((size_t)NNODE * HCDIM * sizeof(float));
    float*  kbuf     = (float*) alloc((size_t)NNODE * HCDIM * sizeof(float));
    float*  vbuf     = (float*) alloc((size_t)NNODE * HCDIM * sizeof(float));
    float*  rbuf     = (float*) alloc((size_t)NNODE * HCDIM * sizeof(float));
    float*  aggbuf   = (float*) alloc((size_t)NNODE * HCDIM * sizeof(float));
    float*  alog     = (float*) alloc((size_t)NEDGE * NHEAD * sizeof(float));
    float*  pbuf     = (float*) alloc((size_t)NEDGE * NHEAD * sizeof(float));
    unsigned* amaxb  = (unsigned*)alloc((size_t)NNODE * NHEAD * sizeof(unsigned));
    float*  den      = (float*) alloc((size_t)NNODE * NHEAD * sizeof(float));
    float*  maskbuf  = (float*) alloc((size_t)NNODE * sizeof(float));
    float*  gacc     = (float*) alloc((size_t)NGRAPH * 2 * DIM * sizeof(float));
    bf16_t* g_b16    = (bf16_t*)alloc((size_t)NGRAPH * 2 * DIM * sizeof(bf16_t));
    float*  h1       = (float*) alloc((size_t)NGRAPH * DNHID * sizeof(float));
    bf16_t* h1_b16   = (bf16_t*)alloc((size_t)NGRAPH * DNHID * sizeof(bf16_t));
    float*  h2       = (float*) alloc((size_t)NGRAPH * (DNHID/2) * sizeof(float));
    bf16_t* h2_b16   = (bf16_t*)alloc((size_t)NGRAPH * (DNHID/2) * sizeof(bf16_t));

    auto cvt = [&](const float* s, bf16_t* d, size_t n) {
        k_cvt_f32_bf16<<<dim3((unsigned)((n + 255) / 256)), dim3(256), 0, stream>>>(s, d, n);
    };
    auto cvtT = [&](const float* s, bf16_t* d, int K, int Nout, int NoutPad) {
        size_t n = (size_t)NoutPad * K;
        k_cvt_transpose_pad<<<dim3((unsigned)((n + 255) / 256)), dim3(256), 0, stream>>>(
            s, d, K, Nout, NoutPad);
    };
    auto fillf = [&](float* p, float v, size_t n) {
        k_fill_f32<<<dim3((unsigned)((n + 255) / 256)), dim3(256), 0, stream>>>(p, v, n);
    };
    auto fillu = [&](unsigned* p, unsigned v, size_t n) {
        k_fill_u32<<<dim3((unsigned)((n + 255) / 256)), dim3(256), 0, stream>>>(p, v, n);
    };
    auto gemm = [&](const bf16_t* A, const bf16_t* Wt, const float* bias, float* C,
                    int M, int K, int Nout, int NoutPad, int ldc, int mode,
                    const float* g_ = nullptr, const float* b_ = nullptr,
                    const float* m_ = nullptr, const float* v_ = nullptr) {
        dim3 grd((M + 63) / 64, NoutPad / 16);
        k_wmma_gemm_bf16<<<grd, dim3(128), 0, stream>>>(A, Wt, bias, C, M, K, Nout, ldc,
                                                        mode, g_, b_, m_, v_);
    };

    // --- convert all GEMM weights to transposed/padded bf16 ---
    for (int l = 0; l < 5; ++l) {
        const float* srcs[4] = {conv[l].Wq, conv[l].Wk, conv[l].Wv, conv[l].Ws};
        for (int j = 0; j < 4; ++j) cvtT(srcs[j], cWb[l][j], cdin[l], HCDIM, HCDIM);
        cvtT(transfW[l], tWb[l], HCDIM, DIM, DIM);
    }
    for (int j = 0; j < 3; ++j) cvtT(headW[j], hWb[j], hK[j], hN[j], hNp[j]);

    // --- state init ---
    fillf(maskbuf, 1.0f, NNODE);
    fillf(gacc, 0.0f, (size_t)NGRAPH * 2 * DIM);
    cvt(x_in, xin_b16, (size_t)NNODE * FIN);

    const int EH = NEDGE * NHEAD;
    dim3 ehgrid((EH + 255) / 256), ehblk(256);

    auto run_conv = [&](int l, const bf16_t* A, int din) {
        const float* cb[4] = {conv[l].bq, conv[l].bk, conv[l].bv, conv[l].bs};
        float* outs[4] = {qbuf, kbuf, vbuf, rbuf};
        for (int j = 0; j < 4; ++j)
            gemm(A, cWb[l][j], cb[j], outs[j], NNODE, din, HCDIM, HCDIM, HCDIM, 0);
        fillu(amaxb, 0u, (size_t)NNODE * NHEAD);
        fillf(den, 0.f, (size_t)NNODE * NHEAD);
        fillf(aggbuf, 0.f, (size_t)NNODE * HCDIM);
        k_edge_logits<<<ehgrid, ehblk, 0, stream>>>(qbuf, kbuf, e_src, e_dst, maskbuf, alog, amaxb);
        k_edge_softmax<<<ehgrid, ehblk, 0, stream>>>(alog, amaxb, e_dst, pbuf, den);
        k_edge_aggregate<<<ehgrid, ehblk, 0, stream>>>(pbuf, den, vbuf, e_src, e_dst, aggbuf);
        k_beta_gate<<<dim3((NNODE + 127) / 128), dim3(128), 0, stream>>>(aggbuf, rbuf,
                                                                         conv[l].Wbeta, xhc_b16);
    };

    // conv1 + transf1 + bn1
    run_conv(0, xin_b16, FIN);
    gemm(xhc_b16, tWb[0], transfB[0], x64, NNODE, HCDIM, DIM, DIM, DIM, 2,
         bng[0], bnb[0], bnm[0], bnv[0]);

    // 4 conv/pool stages
    int kcur = NPG;
    for (int i = 0; i < 4; ++i) {
        cvt(x64, x64_b16, (size_t)NNODE * DIM);
        run_conv(1 + i, x64_b16, DIM);
        gemm(xhc_b16, tWb[1 + i], transfB[1 + i], x64, NNODE, HCDIM, DIM, DIM, DIM, 2,
             bng[1 + i], bnb[1 + i], bnm[1 + i], bnv[1 + i]);
        kcur = (kcur + 1) / 2;  // ratio 0.5, ceil
        k_topk_pool<<<dim3(NGRAPH), dim3(NPG), 0, stream>>>(x64, poolw[i], maskbuf, kcur);
        k_global_pool<<<dim3(NGRAPH), dim3(DIM), 0, stream>>>(x64, maskbuf, gacc, 1.0f / kcur);
    }

    // head MLP (M=256)
    cvt(gacc, g_b16, (size_t)NGRAPH * 2 * DIM);
    gemm(g_b16, hWb[0], headB[0], h1, NGRAPH, 2 * DIM, DNHID, DNHID, DNHID, 1);
    cvt(h1, h1_b16, (size_t)NGRAPH * DNHID);
    gemm(h1_b16, hWb[1], headB[1], h2, NGRAPH, DNHID, DNHID / 2, DNHID / 2, DNHID / 2, 1);
    cvt(h2, h2_b16, (size_t)NGRAPH * (DNHID / 2));
    gemm(h2_b16, hWb[2], headB[2], (float*)d_out, NGRAPH, DNHID / 2, NCLS, 16, NCLS, 0);
}
